// Sequence_21277267984826
// MI455X (gfx1250) — compile-verified, matrix operand order checked by reading, not run on hardware
//
#include <hip/hip_runtime.h>
#include <math.h>

#define H 64
#define WAVES 4
#define ROWS_PER_WAVE 16
#define ROWS_PER_WG (WAVES * ROWS_PER_WAVE)
#define FRAG_ELEMS (32 * 32 * 16)   // 32 frags x 32 lanes x 16 halves per weight matrix

typedef __attribute__((ext_vector_type(16))) _Float16 v16h;
typedef __attribute__((ext_vector_type(8)))  _Float16 v8h;
typedef __attribute__((ext_vector_type(8)))  float    v8f;

static __device__ __forceinline__ float sigmoidf_(float x) {
    return 1.0f / (1.0f + __expf(-x));
}

static __device__ __forceinline__ float tanhf_(float x) {
#if __has_builtin(__builtin_amdgcn_tanhf)
    return __builtin_amdgcn_tanhf(x);   // gfx1250 v_tanh_f32
#else
    return tanhf(x);
#endif
}

// Persistent 2-layer LSTM. One wave owns 16 batch rows for the whole sequence.
// Weights live in LDS as pre-swizzled f16 WMMA B-fragments; c-state lives in the
// WMMA f32 accumulator lane layout; h-state round-trips through per-wave LDS
// (in-order per-wave DS ops -> no barriers inside the time loop).
// Gate tiles are computed 4-at-a-time (i,f,g,o for one h-column tile) so only
// 4 accumulator tiles are ever live -> no VGPR spilling.
__global__ void __launch_bounds__(ROWS_PER_WG * 2, 1)
lstm2_persistent(const float* __restrict__ input,
                 const float* __restrict__ Wih1,
                 const float* __restrict__ Whh1,
                 const float* __restrict__ bih1,
                 const float* __restrict__ bhh1,
                 const float* __restrict__ Wih2,
                 const float* __restrict__ Whh2,
                 const float* __restrict__ bih2,
                 const float* __restrict__ bhh2,
                 const float* __restrict__ Wl,
                 const float* __restrict__ bl,
                 const int*   __restrict__ futurep,
                 float* __restrict__ out,
                 int B, int T)
{
    extern __shared__ char smem[];
    _Float16* Bw1 = (_Float16*)smem;            // Whh1^T fragments (layer1 recurrent)
    _Float16* Bw2 = Bw1 + FRAG_ELEMS;           // Wih2^T fragments (layer2 input)
    _Float16* Bw3 = Bw2 + FRAG_ELEMS;           // Whh2^T fragments (layer2 recurrent)
    _Float16* h1s = Bw3 + FRAG_ELEMS;           // WAVES x 16 x 64 f16
    _Float16* h2s = h1s + WAVES * 16 * H;       // WAVES x 16 x 64 f16
    float*    xs    = (float*)(h2s + WAVES * 16 * H);  // WAVES x 16
    float*    outs  = xs + WAVES * 16;                 // WAVES x 16
    float*    wls   = outs + WAVES * 16;               // 64
    float*    wih1s = wls + H;                         // 256 (gate constants in LDS,
    float*    b1s   = wih1s + 4 * H;                   // 256  not VGPRs: they were
    float*    b2s   = b1s + 4 * H;                     // 256  spilling to scratch)

    const int tid  = threadIdx.x;
    const int lane = tid & 31;
    const int wv   = tid >> 5;
    const int nn   = lane & 15;                 // N column / A row within tile
    const int moff = (lane < 16) ? 0 : 8;       // C/D: lanes 16-31 hold rows M+8

    // ---- one-time preload: f32 weights -> f16 B-fragments, swizzled so that
    // fragment (kt,nt) for lane l is 16 contiguous halves.
    // B[k][n] = W[n][k]; per-lane K mapping mirrors the ISA A-matrix 16x32 layout.
    auto swz = [&](_Float16* dst, const float* W) {
        for (int e = tid; e < FRAG_ELEMS; e += blockDim.x) {
            int f  = e >> 9;            // fragment id 0..31
            int ln = (e >> 4) & 31;     // lane
            int hh = e & 15;            // half index within lane
            int kt = f >> 4, nt = f & 15;
            int kk = ((hh < 8) ? hh : hh + 8) + ((ln < 16) ? 0 : 8);
            int k  = kt * 32 + kk;
            int n  = nt * 16 + (ln & 15);
            dst[e] = (_Float16)W[n * H + k];
        }
    };
    swz(Bw1, Whh1);
    swz(Bw2, Wih2);
    swz(Bw3, Whh2);
    for (int i = tid; i < WAVES * 16 * H; i += blockDim.x) {
        h1s[i] = (_Float16)0.f;
        h2s[i] = (_Float16)0.f;
    }
    for (int i = tid; i < H; i += blockDim.x) wls[i] = Wl[i];
    for (int i = tid; i < 4 * H; i += blockDim.x) {
        wih1s[i] = Wih1[i];
        b1s[i]   = bih1[i] + bhh1[i];
        b2s[i]   = bih2[i] + bhh2[i];
    }
    __syncthreads();   // the only barrier: weights/zeros visible to all waves

    const float blv    = bl[0];
    const int   future = futurep[0];
    const int   b0     = (blockIdx.x * WAVES + wv) * ROWS_PER_WAVE;

    _Float16* h1w = h1s + wv * 16 * H;
    _Float16* h2w = h2s + wv * 16 * H;
    float*    xw  = xs   + wv * 16;
    float*    ow  = outs + wv * 16;

    // cell state in WMMA accumulator layout: tile t covers h-columns 16t+nn,
    // element r covers row moff+r.
    v8f c1[4], c2[4];
#pragma unroll
    for (int t = 0; t < 4; ++t) {
        v8f z = {0.f, 0.f, 0.f, 0.f, 0.f, 0.f, 0.f, 0.f};
        c1[t] = z;
        c2[t] = z;
    }

    // A-fragment load (16x32 f16): lane nn = row; halves K 0..7/16..23 (lanes<16)
    // or 8..15/24..31 (lanes>=16) -> two contiguous 16B LDS loads.
    auto loadA = [&](const _Float16* hs, int kc) -> v16h {
        const _Float16* base = hs + nn * H + kc * 32 + moff;
        v8h lo = *(const v8h*)(base);
        v8h hi = *(const v8h*)(base + 16);
        return __builtin_shufflevector(lo, hi, 0, 1, 2, 3, 4, 5, 6, 7,
                                       8, 9, 10, 11, 12, 13, 14, 15);
    };

    // acc += h(16x64) @ W^T restricted to gate tile nt: 2 WMMAs (K=64)
    auto mm2 = [&](v8f acc, v16h a0, v16h a1, const _Float16* Bsw, int nt) -> v8f {
        v16h bf0 = *(const v16h*)(Bsw + (nt * 32 + lane) * 16);
        acc = __builtin_amdgcn_wmma_f32_16x16x32_f16(
            false, a0, false, bf0, (short)0, acc, false, false);
        v16h bf1 = *(const v16h*)(Bsw + ((16 + nt) * 32 + lane) * 16);
        acc = __builtin_amdgcn_wmma_f32_16x16x32_f16(
            false, a1, false, bf1, (short)0, acc, false, false);
        return acc;
    };

    // layer-1 gate init: x * Wih1 + b1 (rank-1, F==1), constants from LDS
    auto initG1 = [&](const float* xvp, int nt) -> v8f {
        float w = wih1s[nt * 16 + nn];
        float b = b1s[nt * 16 + nn];
        v8f g;
#pragma unroll
        for (int r = 0; r < 8; ++r) g[r] = fmaf(xvp[r], w, b);
        return g;
    };
    // layer-2 gate init: bias broadcast
    auto initG2 = [&](int nt) -> v8f {
        float b = b2s[nt * 16 + nn];
        v8f g;
#pragma unroll
        for (int r = 0; r < 8; ++r) g[r] = b;
        return g;
    };

    // one h-column tile's LSTM cell update (i,f,g,o) + h writeback as f16
    auto cellT = [&](v8f gi, v8f gf, v8f gg, v8f go, v8f& c, _Float16* hdst, int t) {
#pragma unroll
        for (int r = 0; r < 8; ++r) {
            float iv = sigmoidf_(gi[r]);
            float fv = sigmoidf_(gf[r]);
            float gv = tanhf_(gg[r]);
            float ov = sigmoidf_(go[r]);
            float cn = fmaf(fv, c[r], iv * gv);
            c[r] = cn;
            hdst[(r + moff) * H + t * 16 + nn] = (_Float16)(ov * tanhf_(cn));
        }
    };

    auto step = [&](const float* xp) {
        float xv[8];
#pragma unroll
        for (int r = 0; r < 8; ++r) xv[r] = xp[r + moff];

        // ---- layer 1: gates = x*Wih1 + b1 + h1 @ Whh1^T
        {
            v16h a0 = loadA(h1w, 0);   // old h1 (register-resident before writes)
            v16h a1 = loadA(h1w, 1);
#pragma unroll
            for (int t = 0; t < 4; ++t) {
                v8f gi = mm2(initG1(xv, t),      a0, a1, Bw1, t);
                v8f gf = mm2(initG1(xv, 4 + t),  a0, a1, Bw1, 4 + t);
                v8f gg = mm2(initG1(xv, 8 + t),  a0, a1, Bw1, 8 + t);
                v8f go = mm2(initG1(xv, 12 + t), a0, a1, Bw1, 12 + t);
                cellT(gi, gf, gg, go, c1[t], h1w, t);
            }
        }
        // ---- layer 2: gates = b2 + h1 @ Wih2^T + h2 @ Whh2^T
        {
            v16h a0 = loadA(h1w, 0);   // new h1 (per-wave LDS ops are in-order)
            v16h a1 = loadA(h1w, 1);
            v16h a2 = loadA(h2w, 0);   // old h2
            v16h a3 = loadA(h2w, 1);
#pragma unroll
            for (int t = 0; t < 4; ++t) {
                v8f gi = mm2(mm2(initG2(t),      a0, a1, Bw2, t),      a2, a3, Bw3, t);
                v8f gf = mm2(mm2(initG2(4 + t),  a0, a1, Bw2, 4 + t),  a2, a3, Bw3, 4 + t);
                v8f gg = mm2(mm2(initG2(8 + t),  a0, a1, Bw2, 8 + t),  a2, a3, Bw3, 8 + t);
                v8f go = mm2(mm2(initG2(12 + t), a0, a1, Bw2, 12 + t), a2, a3, Bw3, 12 + t);
                cellT(gi, gf, gg, go, c2[t], h2w, t);
            }
        }
    };

    auto readout = [&]() {
        if (lane < 16) {
            float s = blv;
            for (int k = 0; k < H; ++k) s += (float)h2w[lane * H + k] * wls[k];
            ow[lane] = s;
        }
    };

    // ---- main scan over time (sequential dependence; the critical path) ----
    for (int t = 0; t < T; ++t) {
        if (lane < 16) {
            long idx = (long)(b0 + lane) * T + t;
            xw[lane] = input[idx];
            if (t + 64 < T) __builtin_prefetch(input + idx + 64, 0, 1);
        }
        step(xw);
    }
    readout();

    // ---- autoregressive future steps ----
    for (int f = 0; f < future; ++f) {
        step(ow);
        readout();
    }

    if (lane < 16) out[b0 + lane] = ow[lane];
}

extern "C" void kernel_launch(void* const* d_in, const int* in_sizes, int n_in,
                              void* d_out, int out_size, void* d_ws, size_t ws_size,
                              hipStream_t stream) {
    (void)n_in; (void)d_ws; (void)ws_size;
    const float* input = (const float*)d_in[0];
    const float* Wih1  = (const float*)d_in[1];
    const float* Whh1  = (const float*)d_in[2];
    const float* bih1  = (const float*)d_in[3];
    const float* bhh1  = (const float*)d_in[4];
    const float* Wih2  = (const float*)d_in[5];
    const float* Whh2  = (const float*)d_in[6];
    const float* bih2  = (const float*)d_in[7];
    const float* bhh2  = (const float*)d_in[8];
    const float* Wl    = (const float*)d_in[9];
    const float* bl    = (const float*)d_in[10];
    const int*   fut   = (const int*)d_in[11];

    const int B = out_size;               // output is [B,1,1] f32
    const int T = in_sizes[0] / B;        // input is [B,T,1]
    const int grid = (B + ROWS_PER_WG - 1) / ROWS_PER_WG;

    const size_t smem = (size_t)(3 * FRAG_ELEMS + 2 * WAVES * 16 * H) * 2   // f16
                      + (size_t)(2 * WAVES * 16 + H + 3 * 4 * H) * sizeof(float);

    lstm2_persistent<<<grid, WAVES * 32, smem, stream>>>(
        input, Wih1, Whh1, bih1, bhh1, Wih2, Whh2, bih2, bhh2,
        Wl, bl, fut, (float*)d_out, B, T);
}
